// MutiHeadSelfAttention_74741020885218
// MI455X (gfx1250) — matrix operand order repaired and checked by reading, not compiled, not run on hardware
//
#include <hip/hip_runtime.h>

typedef __attribute__((ext_vector_type(16))) _Float16 v16h;
typedef __attribute__((ext_vector_type(8)))  _Float16 v8h;
typedef __attribute__((ext_vector_type(4)))  _Float16 v4h;
typedef __attribute__((ext_vector_type(4)))  float    v4f;
typedef __attribute__((ext_vector_type(8)))  float    v8f;
typedef __attribute__((ext_vector_type(4)))  int      v4i;

#define D_MODEL 1024
#define T_SEQ   2048
#define NHEAD   16
#define DK      64
#define BQ      128
#define BK      32
#define NROWS   4096   // B*T

#if __has_builtin(__builtin_amdgcn_global_load_async_to_lds_b128)
#define HAVE_ASYNC_LDS 1
#else
#define HAVE_ASYNC_LDS 0
#endif

#if HAVE_ASYNC_LDS
#if __has_builtin(__builtin_amdgcn_s_wait_asynccnt)
#define STAGE_WAIT(N) __builtin_amdgcn_s_wait_asynccnt(N)
#else
#define STAGE_WAIT(N) asm volatile("s_wait_asynccnt " #N ::: "memory")
#endif
#else
#define STAGE_WAIT(N) do {} while (0)
#endif

// ---------------------------------------------------------------------------
// WMMA fragment helpers (CDNA5 wave32 layouts, CDNA5 ISA §7.12.2)
// A (16x32 f16): lane L holds row M=L&15; K = {0..7,16..23} (L<16) or
//                {8..15,24..31} (L>=16)  -> two 16B loads.
// B (32x16 f16): lane L holds col N=L&15; K = {0..7,16..23} (L<16) or
//                {8..15,24..31} (L>=16) -> two 16B loads from an N-major tile.
// C/D (16x16 f32): lane L, vgpr r -> (M = r + 8*(L>=16), N = L&15).
// ---------------------------------------------------------------------------

static __device__ __forceinline__ v16h load_a16x32(const _Float16* base, int stride) {
  const int lane = threadIdx.x & 31;
  const int m  = lane & 15;
  const int kb = (lane >> 4) << 3;      // 0 or 8
  const _Float16* p = base + m * stride + kb;
  v8h lo = *(const v8h*)(p);            // K = kb .. kb+7
  v8h hi = *(const v8h*)(p + 16);       // K = 16+kb .. 16+kb+7
  v16h r;
#pragma unroll
  for (int i = 0; i < 8; ++i) { r[i] = lo[i]; r[i + 8] = hi[i]; }
  return r;
}

static __device__ __forceinline__ v16h load_b32x16(const _Float16* base, int stride) {
  const int lane = threadIdx.x & 31;
  const int n  = lane & 15;
  const int kb = (lane >> 4) << 3;      // 0 or 8
  const _Float16* p = base + n * stride + kb;
  v8h lo = *(const v8h*)(p);
  v8h hi = *(const v8h*)(p + 16);
  v16h r;
#pragma unroll
  for (int i = 0; i < 8; ++i) { r[i] = lo[i]; r[i + 8] = hi[i]; }
  return r;
}

static __device__ __forceinline__ v8f wmma_f16(v16h a, v16h b, v8f c) {
  return __builtin_amdgcn_wmma_f32_16x16x32_f16(false, a, false, b, (short)0, c,
                                                false, false);
}

static __device__ __forceinline__ float redmax16(float v) {
#pragma unroll
  for (int m = 1; m < 16; m <<= 1) v = fmaxf(v, __shfl_xor(v, m, 32));
  return v;
}
static __device__ __forceinline__ float redsum16(float v) {
#pragma unroll
  for (int m = 1; m < 16; m <<= 1) v += __shfl_xor(v, m, 32);
  return v;
}

// 32 bytes (16 f16) global -> LDS, async (ASYNCcnt) if available.
static __device__ __forceinline__ void stage32B(const _Float16* g, _Float16* l) {
#if HAVE_ASYNC_LDS
  __attribute__((address_space(1))) v4i* gp =
      (__attribute__((address_space(1))) v4i*)(unsigned long long)(size_t)g;
  __attribute__((address_space(3))) v4i* lp =
      (__attribute__((address_space(3))) v4i*)(unsigned int)(size_t)l;
  // IOFFSET is added to BOTH the LDS and global address (ISA 10.x async ops)
  __builtin_amdgcn_global_load_async_to_lds_b128(gp, lp, 0, 0);
  __builtin_amdgcn_global_load_async_to_lds_b128(gp, lp, 16, 0);
#else
  *(v8h*)l       = *(const v8h*)g;
  *(v8h*)(l + 8) = *(const v8h*)(g + 8);
#endif
}

// ---------------------------------------------------------------------------
// fp32 -> f16 conversion, vectorized x4
// ---------------------------------------------------------------------------
__global__ __launch_bounds__(256) void k_cvt4(const float* __restrict__ src,
                                              _Float16* __restrict__ dst, int n4) {
  int i = blockIdx.x * 256 + threadIdx.x;
  if (i < n4) {
    v4f f = ((const v4f*)src)[i];
    v4h o;
#pragma unroll
    for (int k = 0; k < 4; ++k) o[k] = (_Float16)f[k];
    ((v4h*)dst)[i] = o;
  }
}

// ---------------------------------------------------------------------------
// 32(M) x 64(N) x 1024(K) WMMA GEMM core, ping-pong double buffering (no
// cross-buffer register copies).  Abase: token rows (stride D_MODEL, two
// 16-row subtiles).  Wbase: output-feature rows (stride D_MODEL, 4 n-tiles).
// ---------------------------------------------------------------------------
#define GEMM_LOAD(slot, kk)                                                    \
  do {                                                                         \
    a0[slot] = load_a16x32(Abase + (kk), D_MODEL);                             \
    a1[slot] = load_a16x32(Abase + 16 * D_MODEL + (kk), D_MODEL);              \
    _Pragma("unroll") for (int j = 0; j < 4; ++j) bb[slot][j] =                \
        load_b32x16(Wbase + (size_t)(16 * j) * D_MODEL + (kk), D_MODEL);       \
  } while (0)

#define GEMM_COMP(slot)                                                        \
  do {                                                                         \
    _Pragma("unroll") for (int j = 0; j < 4; ++j) {                            \
      acc[0][j] = wmma_f16(a0[slot], bb[slot][j], acc[0][j]);                  \
      acc[1][j] = wmma_f16(a1[slot], bb[slot][j], acc[1][j]);                  \
    }                                                                          \
  } while (0)

static __device__ __forceinline__ void gemm_32x64(const _Float16* __restrict__ Abase,
                                                  const _Float16* __restrict__ Wbase,
                                                  v8f (&acc)[2][4]) {
  v16h a0[2], a1[2], bb[2][4];
  GEMM_LOAD(0, 0);
  GEMM_LOAD(1, 32);
#pragma unroll 1
  for (int kk = 0; kk + 96 < D_MODEL; kk += 64) {
    GEMM_COMP(0);
    GEMM_LOAD(0, kk + 64);
    GEMM_COMP(1);
    GEMM_LOAD(1, kk + 96);
  }
  GEMM_COMP(0);
  GEMM_COMP(1);
}

// ---------------------------------------------------------------------------
// QKV projection (+RoPE on Q,K).  y[t,e] = sum_d x[t,d] * W[e,d]
// blockIdx.x: 256-token tile, blockIdx.y: head, blockIdx.z: 0=Q 1=K 2=V
// ---------------------------------------------------------------------------
__global__ __launch_bounds__(256) void k_qkv_rope(
    const _Float16* __restrict__ Xh,
    const _Float16* __restrict__ Wqh, const _Float16* __restrict__ Wkh,
    const _Float16* __restrict__ Wvh,
    _Float16* __restrict__ Qh, _Float16* __restrict__ Kh,
    _Float16* __restrict__ Vt) {
  const int wid  = threadIdx.x >> 5;
  const int lane = threadIdx.x & 31;
  const int row0 = blockIdx.x * 256 + wid * 32;   // token-row tile in [0,4096)
  const int h    = blockIdx.y;
  const int which = blockIdx.z;
  const _Float16* W = (which == 0) ? Wqh : (which == 1) ? Wkh : Wvh;

  v8f acc[2][4] = {};
  gemm_32x64(Xh + (size_t)row0 * D_MODEL, W + (size_t)(h * DK) * D_MODEL, acc);

  const int n  = lane & 15;
  const int hi = lane >> 4;
  if (which == 2) {
    // V: transposed layout [B,H,dk,T]; a lane's 8 acc rows are 8 consecutive t
#pragma unroll
    for (int s = 0; s < 2; ++s) {
      const int rowb = row0 + 16 * s + 8 * hi;
      const int b = rowb >> 11, t = rowb & (T_SEQ - 1);
#pragma unroll
      for (int j = 0; j < 4; ++j) {
        const int d = 16 * j + n;
        v8h pv;
#pragma unroll
        for (int r = 0; r < 8; ++r) pv[r] = (_Float16)acc[s][j][r];
        *(v8h*)&Vt[((size_t)(b * NHEAD + h) * DK + d) * T_SEQ + t] = pv;
      }
    }
  } else {
    _Float16* Out = (which == 0) ? Qh : Kh;
#pragma unroll
    for (int s = 0; s < 2; ++s) {
#pragma unroll
      for (int j = 0; j < 4; ++j) {
        const int d = 16 * j + n;                 // dim within head
        // inv_freq = 10000^{-(d/2)/32}
        const float invf = __expf(-(float)(d >> 1) * (9.2103403719761836f / 32.0f));
#pragma unroll
        for (int r = 0; r < 8; ++r) {
          const int row = row0 + 16 * s + r + 8 * hi;
          const int b = row >> 11, t = row & (T_SEQ - 1);
          float v = acc[s][j][r];
          float partner = __shfl_xor(v, 1, 32);   // RoPE pair = adjacent lane
          float sn, cs;
          __sincosf((float)t * invf, &sn, &cs);
          float o = (d & 1) ? (v * cs + partner * sn)
                            : (v * cs - partner * sn);
          Out[((size_t)(b * NHEAD + h) * T_SEQ + t) * DK + d] = (_Float16)o;
        }
      }
    }
  }
}

// ---------------------------------------------------------------------------
// Flash attention, causal.  Block = 8 waves = 128 query rows of one (b,h).
// Double-buffered K/V^T tiles staged with async-to-LDS (ASYNCcnt); next tile's
// loads issued before computing the current one, waited with asynccnt<=2.
// ---------------------------------------------------------------------------
__global__ __launch_bounds__(256) void k_attn(const _Float16* __restrict__ Qh,
                                              const _Float16* __restrict__ Kh,
                                              const _Float16* __restrict__ Vt,
                                              _Float16* __restrict__ Ah) {
  __shared__ _Float16 sK[2][BK * DK];    // [32 keys][64]
  __shared__ _Float16 sV[2][DK * BK];    // [64 dims][32 keys]  (V^T tile)
  __shared__ _Float16 sP[8][16 * BK];    // per-wave P tile [16][32]

  const int wid  = threadIdx.x >> 5;
  const int lane = threadIdx.x & 31;
  const int tid  = threadIdx.x;
  const int q0   = blockIdx.x * BQ;
  const int h    = blockIdx.y;
  const int b    = blockIdx.z;
  const size_t head = (size_t)(b * NHEAD + h);
  const _Float16* Qb = Qh + head * T_SEQ * DK;
  const _Float16* Kb = Kh + head * T_SEQ * DK;
  const _Float16* Vb = Vt + head * DK * T_SEQ;
  const int qrow = q0 + wid * 16;

  // per-thread staging role (K tile: threads 0..127, V^T tile: 128..255)
  const int krow = tid >> 2,          kcol = (tid & 3) * 16;
  const int vrow = (tid - 128) >> 1,  vcol = ((tid - 128) & 1) * 16;

  // Q fragments, pre-scaled by 1/sqrt(dk) = 1/8 (exact in f16)
  v16h aq0 = load_a16x32(Qb + (size_t)qrow * DK, DK);
  v16h aq1 = load_a16x32(Qb + (size_t)qrow * DK + 32, DK);
#pragma unroll
  for (int i = 0; i < 16; ++i) {
    aq0[i] = aq0[i] * (_Float16)0.125f;
    aq1[i] = aq1[i] * (_Float16)0.125f;
  }

  float mrow[8], lrow[8];
  v8f acc[4] = {};
#pragma unroll
  for (int r = 0; r < 8; ++r) { mrow[r] = -1e30f; lrow[r] = 0.0f; }

  const int kend = q0 + BQ;   // max key needed by this block (causal)

  // stage tile 0 into buffer 0
  if (tid < 128) stage32B(Kb + (size_t)krow * DK + kcol, &sK[0][krow * DK + kcol]);
  else           stage32B(Vb + (size_t)vrow * T_SEQ + vcol, &sV[0][vrow * BK + vcol]);

#pragma unroll 1
  for (int kk = 0; kk < kend; kk += BK) {
    const int cur = (kk >> 5) & 1;
    if (kk + BK < kend) {
      // issue next tile's async loads into the other buffer, then wait for
      // the CURRENT tile only (async loads complete in order; <=2 leaves the
      // two just-issued loads of each thread in flight)
      const int nxt = cur ^ 1;
      if (tid < 128)
        stage32B(Kb + (size_t)(kk + BK + krow) * DK + kcol, &sK[nxt][krow * DK + kcol]);
      else
        stage32B(Vb + (size_t)vrow * T_SEQ + (kk + BK) + vcol, &sV[nxt][vrow * BK + vcol]);
      STAGE_WAIT(2);
    } else {
      STAGE_WAIT(0);
    }
    __syncthreads();

    // ---- S = Q K^T : two 16-key tiles, K-dim 64 = 2 WMMAs each ----
    const _Float16* cK = sK[cur];
    const _Float16* cV = sV[cur];
    v8f z = {};
    v16h b00 = load_b32x16(cK, DK);             // keys 0..15,  dk 0..31
    v16h b01 = load_b32x16(cK + 32, DK);        // keys 0..15,  dk 32..63
    v16h b10 = load_b32x16(cK + 16 * DK, DK);   // keys 16..31, dk 0..31
    v16h b11 = load_b32x16(cK + 16 * DK + 32, DK);
    v8f s0 = wmma_f16(aq0, b00, z);
    s0 = wmma_f16(aq1, b01, s0);
    v8f s1 = wmma_f16(aq0, b10, z);
    s1 = wmma_f16(aq1, b11, s1);

    // ---- causal mask + online softmax ----
    const int n  = lane & 15;
    const int hi = lane >> 4;
#pragma unroll
    for (int r = 0; r < 8; ++r) {
      const int m  = r + 8 * hi;
      const int rq = qrow + m;
      float v0 = (kk + n      <= rq) ? s0[r] : -1e30f;
      float v1 = (kk + 16 + n <= rq) ? s1[r] : -1e30f;
      float mx = redmax16(fmaxf(v0, v1));
      const float mnew  = fmaxf(mrow[r], mx);
      const float scale = __expf(mrow[r] - mnew);
      const float p0 = __expf(v0 - mnew);
      const float p1 = __expf(v1 - mnew);
      lrow[r] = lrow[r] * scale + redsum16(p0 + p1);
      mrow[r] = mnew;
#pragma unroll
      for (int j = 0; j < 4; ++j) acc[j][r] *= scale;
      sP[wid][m * BK + n]      = (_Float16)p0;   // C-layout -> A-layout via LDS
      sP[wid][m * BK + 16 + n] = (_Float16)p1;
    }
    __syncthreads();

    // ---- O += P V : A = P (16x32), B = V^T tile slices (N-major) ----
    v16h ap = load_a16x32(&sP[wid][0], BK);
#pragma unroll
    for (int j = 0; j < 4; ++j) {
      v16h bv = load_b32x16(cV + (16 * j) * BK, BK);
      acc[j] = wmma_f16(ap, bv, acc[j]);
    }
    __syncthreads();   // releases this buffer for restaging next iteration
  }

  // ---- epilogue: normalize, write [B,T,D] with heads re-interleaved ----
  const int n  = lane & 15;
  const int hi = lane >> 4;
#pragma unroll
  for (int j = 0; j < 4; ++j) {
#pragma unroll
    for (int r = 0; r < 8; ++r) {
      const int m = r + 8 * hi;
      const int q = qrow + m;
      const float o = acc[j][r] / lrow[r];
      Ah[((size_t)(b * T_SEQ + q)) * D_MODEL + h * DK + 16 * j + n] = (_Float16)o;
    }
  }
}

// ---------------------------------------------------------------------------
// Output projection: out[t,e] = sum_d A[t,d] * Wo[e,d], f32 output.
// ---------------------------------------------------------------------------
__global__ __launch_bounds__(256) void k_outproj(const _Float16* __restrict__ Ah,
                                                 const _Float16* __restrict__ Woh,
                                                 float* __restrict__ out) {
  const int wid  = threadIdx.x >> 5;
  const int lane = threadIdx.x & 31;
  const int row0 = blockIdx.x * 256 + wid * 32;
  const int e0   = blockIdx.y * 64;

  v8f acc[2][4] = {};
  gemm_32x64(Ah + (size_t)row0 * D_MODEL, Woh + (size_t)e0 * D_MODEL, acc);

  const int n  = lane & 15;
  const int hi = lane >> 4;
#pragma unroll
  for (int s = 0; s < 2; ++s)
#pragma unroll
    for (int j = 0; j < 4; ++j)
#pragma unroll
      for (int r = 0; r < 8; ++r) {
        const int m = r + 8 * hi;
        out[(size_t)(row0 + 16 * s + m) * D_MODEL + e0 + 16 * j + n] =
            acc[s][j][r];
      }
}

// ---------------------------------------------------------------------------
extern "C" void kernel_launch(void* const* d_in, const int* in_sizes, int n_in,
                              void* d_out, int out_size, void* d_ws, size_t ws_size,
                              hipStream_t stream) {
  const float* x  = (const float*)d_in[0];
  const float* wq = (const float*)d_in[1];
  const float* wk = (const float*)d_in[2];
  const float* wv = (const float*)d_in[3];
  const float* wo = (const float*)d_in[4];

  const size_t nX = (size_t)NROWS * D_MODEL;       // 4,194,304
  const size_t nW = (size_t)D_MODEL * D_MODEL;     // 1,048,576

  // f16 workspace layout (total 48 MiB)
  _Float16* ws  = (_Float16*)d_ws;
  _Float16* Xh  = ws;  ws += nX;
  _Float16* Wqh = ws;  ws += nW;
  _Float16* Wkh = ws;  ws += nW;
  _Float16* Wvh = ws;  ws += nW;
  _Float16* Woh = ws;  ws += nW;
  _Float16* Qh  = ws;  ws += nX;
  _Float16* Kh  = ws;  ws += nX;
  _Float16* Vt  = ws;  ws += nX;
  _Float16* Ah  = ws;  ws += nX;

  k_cvt4<<<(int)(nX / 1024), 256, 0, stream>>>(x,  Xh,  (int)(nX / 4));
  k_cvt4<<<(int)(nW / 1024), 256, 0, stream>>>(wq, Wqh, (int)(nW / 4));
  k_cvt4<<<(int)(nW / 1024), 256, 0, stream>>>(wk, Wkh, (int)(nW / 4));
  k_cvt4<<<(int)(nW / 1024), 256, 0, stream>>>(wv, Wvh, (int)(nW / 4));
  k_cvt4<<<(int)(nW / 1024), 256, 0, stream>>>(wo, Woh, (int)(nW / 4));

  k_qkv_rope<<<dim3(NROWS / 256, NHEAD, 3), 256, 0, stream>>>(Xh, Wqh, Wkh, Wvh,
                                                              Qh, Kh, Vt);
  k_attn<<<dim3(T_SEQ / BQ, NHEAD, 2), 256, 0, stream>>>(Qh, Kh, Vt, Ah);
  k_outproj<<<dim3(NROWS / 256, D_MODEL / 64, 1), 256, 0, stream>>>(Ah, Woh,
                                                                    (float*)d_out);
}
